// TrackPreFilter_88055419502780
// MI455X (gfx1250) — matrix operand order repaired and checked by reading, not compiled, not run on hardware
//
#include <hip/hip_runtime.h>

// Problem dims (from reference)
#define Bn 128
#define Nn 512
#define Cc 7
#define Hh 64
#define Kk 16
#define Rr 2

typedef __bf16 bf16_t;
typedef __attribute__((ext_vector_type(16))) __bf16 v16bf;
typedef __attribute__((ext_vector_type(8)))  __bf16 v8bf;
typedef __attribute__((ext_vector_type(8)))  float  v8f;

typedef __attribute__((ext_vector_type(4))) unsigned int u32x4;
typedef __attribute__((ext_vector_type(4))) int          i32x4;
typedef __attribute__((ext_vector_type(8))) int          i32x8;

// ---------------------------------------------------------------------------
// WMMA fragment helpers (bf16, 16x16x32). h is stored point-major:
// h[b*N*H + point*H + feat], so a point's 64 features are contiguous.
// A fragment (16 rows = points/weights, K=32 chunk):
//   lane<16 holds K {0..15}, lane>=16 holds K {16..31} of row lane&15.
// B fragment (K=32 chunk x 16 cols = points):
//   lane holds 16 contiguous K values of column lane&15 (lo half K0..15, hi K16..31).
// ---------------------------------------------------------------------------
static __device__ __forceinline__ v16bf load_frag_a_g(const bf16_t* base, int ldm, int lane) {
    const int off = (lane & 16) ? 16 : 0;
    const bf16_t* rp = base + (size_t)(lane & 15) * ldm + off;
    union { v16bf v; v8bf h[2]; } u;
    u.h[0] = *(const v8bf*)(rp);
    u.h[1] = *(const v8bf*)(rp + 8);
    return u.v;
}

static __device__ __forceinline__ v16bf load_frag_b_g(const bf16_t* base, int ldn, int lane) {
    const bf16_t* cp = base + (size_t)(lane & 15) * ldn + ((lane & 16) ? 16 : 0);
    return *(const v16bf*)(cp);
}

static __device__ __forceinline__ v8f wmma_bf16(v16bf a, v16bf b, v8f c) {
    return __builtin_amdgcn_wmma_f32_16x16x32_bf16(false, a, false, b, (short)0, c,
                                                   false, false);
}

// ---------------------------------------------------------------------------
// Tensor Data Mover: 1-D contiguous tile (nelem bf16 elements) global -> LDS.
// Descriptor packing per CDNA5 ISA:
//   group0: count=1 | lds_addr | global_addr[56:0] | type=2
//   group1: data_size=1(2B), tensor_dim0=tile_dim0=nelem, tensor_dim1=1,
//           tensor_dim0_stride=nelem, no padding / iterate / multicast.
// Issued by one wave; completion tracked by TENSORcnt.
// ---------------------------------------------------------------------------
static __device__ __forceinline__ void tdm_load_to_lds_1d(const void* gsrc, void* lds_dst,
                                                          unsigned nelem) {
    const unsigned long long ga = (unsigned long long)(size_t)gsrc;
    const unsigned lds = (unsigned)(size_t)lds_dst;

    u32x4 g0;
    g0.x = 1u;                                         // count=1, user descriptor
    g0.y = lds;                                        // lds_addr (bytes)
    g0.z = (unsigned)ga;                               // global_addr[31:0]
    g0.w = ((unsigned)(ga >> 32) & 0x01FFFFFFu)        // global_addr[56:32]
           | (2u << 30);                               // type = 2 ("image")

    i32x8 g1;
    g1[0] = (int)(1u << 16);                           // data_size code 1 => 2 bytes
    g1[1] = (int)((nelem & 0xFFFFu) << 16);            // tensor_dim0[15:0]  @ bits 63:48
    g1[2] = (int)(((nelem >> 16) & 0xFFFFu)            // tensor_dim0[31:16] @ bits 79:64
                  | (1u << 16));                       // tensor_dim1 = 1    @ bits 95:80
    g1[3] = (int)((nelem & 0xFFFFu) << 16);            // tile_dim0          @ bits 127:112
    g1[4] = 0;                                         // tile_dim1 = tile_dim2 = 0 (unused)
    g1[5] = (int)nelem;                                // tensor_dim0_stride[31:0]
    g1[6] = 0;
    g1[7] = 0;

    i32x4 gz = {};
#if __has_include(<hip/amd_detail/amd_gfx1250_TDM.h>)
    // amdgpu-toolchain (clang-23 / therock-10.0 headers): 6-arg form
    i32x8 gz8 = {};
    __builtin_amdgcn_tensor_load_to_lds(g0, g1, gz, gz, gz8, 0);
#else
    // ROCm 7.2 (clang-22): 5-arg form
    __builtin_amdgcn_tensor_load_to_lds(g0, g1, gz, gz, 0);
#endif
}

// ---------------------------------------------------------------------------
// Stem: h = relu(bn(W2 @ relu(bn(W1 @ x))));  write bf16 h and per-point |h|^2
// ---------------------------------------------------------------------------
__global__ __launch_bounds__(256) void stem_kernel(
    const float* __restrict__ x,
    const float* __restrict__ w1, const float* __restrict__ g1, const float* __restrict__ b1,
    const float* __restrict__ w2, const float* __restrict__ g2, const float* __restrict__ b2,
    bf16_t* __restrict__ h_out, float* __restrict__ sq_out) {
    __shared__ float w2s[Hh * Hh];
    __shared__ float w1s[Hh * Cc];
    const int tid = threadIdx.x;
    for (int i = tid; i < Hh * Hh; i += 256) w2s[i] = w2[i];
    for (int i = tid; i < Hh * Cc; i += 256) w1s[i] = w1[i];
    __syncthreads();

    const int gid = blockIdx.x * 256 + tid;      // point id over B*N
    if (gid >= Bn * Nn) return;
    const int b = gid / Nn, n = gid % Nn;

    float xv[Cc];
#pragma unroll
    for (int c = 0; c < Cc; ++c) xv[c] = x[((size_t)b * Cc + c) * Nn + n];

    const float s = rsqrtf(1.0f + 1e-5f);
    float h1[Hh];
    for (int o = 0; o < Hh; ++o) {
        float a = 0.f;
#pragma unroll
        for (int c = 0; c < Cc; ++c) a += w1s[o * Cc + c] * xv[c];
        h1[o] = fmaxf(a * (g1[o] * s) + b1[o], 0.0f);
    }
    float sqa = 0.f;
    bf16_t* hp = h_out + (size_t)gid * Hh;
    for (int o = 0; o < Hh; ++o) {
        float a = 0.f;
        for (int k = 0; k < Hh; ++k) a += w2s[o * Hh + k] * h1[k];
        float v = fmaxf(a * (g2[o] * s) + b2[o], 0.0f);
        bf16_t bv = (bf16_t)v;
        hp[o] = bv;
        float vf = (float)bv;
        sqa += vf * vf;
    }
    sq_out[gid] = sqa;
}

// ---------------------------------------------------------------------------
// kNN: block = (batch, 16-row tile). Inner products via WMMA bf16,
// d_rank = sq_j - 2*inner (sq_i is constant per row). Top-16 per row via
// packed (monotonic-f32-key << 32 | col) wave-min; ties pick the lower index.
// ---------------------------------------------------------------------------
__global__ __launch_bounds__(128) void knn_kernel(
    const bf16_t* __restrict__ h, const float* __restrict__ sq, int* __restrict__ idx_out) {
    __shared__ float dist[16][Nn];               // 32 KB

    const int b    = blockIdx.x >> 5;            // 32 row-tiles per batch
    const int i0   = (blockIdx.x & 31) * 16;
    const int lane = threadIdx.x & 31;
    const int wave = threadIdx.x >> 5;           // 0..3

    const bf16_t* hb  = h  + (size_t)b * Nn * Hh;
    const float*  sqb = sq + (size_t)b * Nn;

    const v16bf a0 = load_frag_a_g(hb + (size_t)i0 * Hh +  0, Hh, lane);
    const v16bf a1 = load_frag_a_g(hb + (size_t)i0 * Hh + 32, Hh, lane);

    for (int jt = 0; jt < 8; ++jt) {             // 4 waves x 8 tiles = 512 cols
        const int j0 = (wave * 8 + jt) * 16;
        // prefetch next tile's column data into cache while this one computes
        if (jt + 1 < 8) {
            const bf16_t* nx = hb + (size_t)(j0 + 16) * Hh + (size_t)(lane & 15) * Hh;
            __builtin_prefetch(nx, 0, 3);
        }
        const v16bf bf0 = load_frag_b_g(hb + (size_t)j0 * Hh +  0, Hh, lane);
        const v16bf bf1 = load_frag_b_g(hb + (size_t)j0 * Hh + 32, Hh, lane);
        v8f c = {};
        c = wmma_bf16(a0, bf0, c);
        c = wmma_bf16(a1, bf1, c);
        const int col = j0 + (lane & 15);
        const float sj = sqb[col];
        const int rhi = (lane & 16) ? 8 : 0;
#pragma unroll
        for (int d = 0; d < 8; ++d)
            dist[rhi + d][col] = sj - 2.0f * c[d];
    }
    __syncthreads();

    // top-16 selection: 4 rows per wave, each lane owns 16 columns
    for (int rr = 0; rr < 4; ++rr) {
        const int row = wave * 4 + rr;
        float v[16];
#pragma unroll
        for (int t = 0; t < 16; ++t) v[t] = dist[row][lane + 32 * t];

        for (int s = 0; s < Kk; ++s) {
            unsigned long long best = ~0ull;
#pragma unroll
            for (int t = 0; t < 16; ++t) {
                unsigned u = __float_as_uint(v[t]);
                u = (u & 0x80000000u) ? ~u : (u | 0x80000000u);
                unsigned long long key =
                    ((unsigned long long)u << 32) | (unsigned)(lane + 32 * t);
                if (key < best) best = key;
            }
            // wave-wide min of 64-bit keys (two 32-bit shuffles)
#pragma unroll
            for (int m = 1; m < 32; m <<= 1) {
                unsigned hi = __shfl_xor((unsigned)(best >> 32), m, 32);
                unsigned lo = __shfl_xor((unsigned)(best & 0xffffffffu), m, 32);
                unsigned long long o = ((unsigned long long)hi << 32) | lo;
                if (o < best) best = o;
            }
            const unsigned win = (unsigned)(best & 0xffffffffu);   // column index
            if (lane == 0)
                idx_out[((size_t)b * Nn + i0 + row) * Kk + s] = (int)win;
#pragma unroll
            for (int t = 0; t < 16; ++t)
                if ((win >> 5) == (unsigned)t && (win & 31u) == (unsigned)lane)
                    v[t] = __builtin_inff();
        }
    }
}

// ---------------------------------------------------------------------------
// EdgeConv: block = (batch, 64 points). Stage h[b] (64 KB bf16) into LDS with
// a single TDM tensor_load_to_lds DMA (TENSORcnt); hold W (64x128) as 16
// register A-fragments; one wave per point builds the 128x16 edge matrix as
// four B fragments and runs 16 WMMAs; BN+ReLU+max over K=16 columns via lane
// shuffles. Writes bf16 h and |h|^2.
// ---------------------------------------------------------------------------
__global__ __launch_bounds__(256) void edgeconv_kernel(
    const bf16_t* __restrict__ h_in, const int* __restrict__ idx,
    const float* __restrict__ w,                      // 64 x 128 (fp32)
    const float* __restrict__ g, const float* __restrict__ bb,
    bf16_t* __restrict__ h_out, float* __restrict__ sq_out) {
    __shared__ bf16_t hlds[Nn * Hh];                  // exactly 64 KB

    const int b    = blockIdx.x >> 3;                 // 8 blocks per batch
    const int pt0  = (blockIdx.x & 7) * 64;
    const int tid  = threadIdx.x;
    const int lane = tid & 31;
    const int wave = tid >> 5;

    // Stage h[b] into LDS via the Tensor Data Mover (one wave issues the DMA).
    const bf16_t* hb = h_in + (size_t)b * Nn * Hh;
    if (wave == 0) {
        tdm_load_to_lds_1d(hb, hlds, Nn * Hh);        // 32768 x 2B = 64 KB
        __builtin_amdgcn_s_wait_tensorcnt(0);
    }
    __syncthreads();

    const int off  = (lane & 16) ? 16 : 0;            // A-fragment half select
    const int koff = (lane & 16) ? 16 : 0;            // B-fragment half select
    const int rhi  = (lane & 16) ? 8 : 0;

    // Hoist W as 16 A-fragments (weight reuse across all points of the block)
    v16bf afrag[4][4];
#pragma unroll
    for (int t = 0; t < 4; ++t) {
        const float* wrow = w + (size_t)(t * 16 + (lane & 15)) * (2 * Hh);
#pragma unroll
        for (int c = 0; c < 4; ++c) {
            union { v16bf v; bf16_t e[16]; } ua;
#pragma unroll
            for (int k = 0; k < 8; ++k) {
                ua.e[k]     = (bf16_t)wrow[c * 32 + off + k];
                ua.e[8 + k] = (bf16_t)wrow[c * 32 + 8 + off + k];
            }
            afrag[t][c] = ua.v;
        }
    }
    // BN scale/bias for the rows this lane produces
    const float s1 = rsqrtf(1.0f + 1e-5f);
    float sc[4][8], bi[4][8];
#pragma unroll
    for (int t = 0; t < 4; ++t)
#pragma unroll
        for (int d = 0; d < 8; ++d) {
            const int m = t * 16 + rhi + d;
            sc[t][d] = g[m] * s1;
            bi[t][d] = bb[m];
        }

    for (int pp = wave; pp < 64; pp += 8) {
        const int p = pt0 + pp;
        const int nj = idx[((size_t)b * Nn + p) * Kk + (lane & 15)]; // this lane's column
        const bf16_t* hc = hlds + (size_t)p  * Hh;
        const bf16_t* hn = hlds + (size_t)nj * Hh;

        // E rows 0..63: center broadcast (same for every column)
        const v16bf b0 = *(const v16bf*)(hc + 0  + koff);
        const v16bf b1 = *(const v16bf*)(hc + 32 + koff);
        // E rows 64..127: neighbor - center (per column gather from LDS)
        union { v16bf v; bf16_t e[16]; } ub2, ub3;
#pragma unroll
        for (int k = 0; k < 16; ++k) {
            ub2.e[k] = (bf16_t)((float)hn[koff + k]      - (float)hc[koff + k]);
            ub3.e[k] = (bf16_t)((float)hn[32 + koff + k] - (float)hc[32 + koff + k]);
        }

        float sqacc = 0.0f;
#pragma unroll
        for (int t = 0; t < 4; ++t) {
            v8f c = {};
            c = wmma_bf16(afrag[t][0], b0,    c);
            c = wmma_bf16(afrag[t][1], b1,    c);
            c = wmma_bf16(afrag[t][2], ub2.v, c);
            c = wmma_bf16(afrag[t][3], ub3.v, c);
#pragma unroll
            for (int d = 0; d < 8; ++d) {
                float val = fmaxf(c[d] * sc[t][d] + bi[t][d], 0.0f);
                // max over the 16 neighbor columns (xor masks stay inside a half)
                val = fmaxf(val, __shfl_xor(val, 8, 32));
                val = fmaxf(val, __shfl_xor(val, 4, 32));
                val = fmaxf(val, __shfl_xor(val, 2, 32));
                val = fmaxf(val, __shfl_xor(val, 1, 32));
                if ((lane & 15) == 0) {
                    const int m = t * 16 + rhi + d;
                    h_out[((size_t)b * Nn + p) * Hh + m] = (bf16_t)val;
                    sqacc += val * val;
                }
            }
        }
        sqacc += __shfl_xor(sqacc, 16, 32);           // combine rows 0..7 / 8..15 halves
        if (lane == 0) sq_out[(size_t)b * Nn + p] = sqacc;
    }
}

// ---------------------------------------------------------------------------
// Head: logits = W_s2 @ relu(bn(W_s1 @ h)) + b_s2
// ---------------------------------------------------------------------------
__global__ __launch_bounds__(256) void head_kernel(
    const bf16_t* __restrict__ h,
    const float* __restrict__ w1, const float* __restrict__ g1, const float* __restrict__ b1,
    const float* __restrict__ w2, const float* __restrict__ b2,
    float* __restrict__ out) {
    __shared__ float wls[Hh * Hh];
    const int tid = threadIdx.x;
    for (int i = tid; i < Hh * Hh; i += 256) wls[i] = w1[i];
    __syncthreads();

    const int gid = blockIdx.x * 256 + tid;
    if (gid >= Bn * Nn) return;

    float hv[Hh];
    const bf16_t* hp = h + (size_t)gid * Hh;
    for (int k = 0; k < Hh; ++k) hv[k] = (float)hp[k];

    const float s = rsqrtf(1.0f + 1e-5f);
    float acc = b2[0];
    for (int o = 0; o < Hh; ++o) {
        float a = 0.f;
        for (int k = 0; k < Hh; ++k) a += wls[o * Hh + k] * hv[k];
        acc += w2[o] * fmaxf(a * (g1[o] * s) + b1[o], 0.0f);
    }
    out[gid] = acc;                                   // (B,1,N) flat = B*N
}

// ---------------------------------------------------------------------------
extern "C" void kernel_launch(void* const* d_in, const int* in_sizes, int n_in,
                              void* d_out, int out_size, void* d_ws, size_t ws_size,
                              hipStream_t stream) {
    (void)in_sizes; (void)n_in; (void)out_size; (void)ws_size;

    const float* x    = (const float*)d_in[0];
    const float* w_t1 = (const float*)d_in[1];
    const float* g_t1 = (const float*)d_in[2];
    const float* b_t1 = (const float*)d_in[3];
    const float* w_t2 = (const float*)d_in[4];
    const float* g_t2 = (const float*)d_in[5];
    const float* b_t2 = (const float*)d_in[6];
    const float* w_nb = (const float*)d_in[7];   // (R, 64, 128)
    const float* g_nb = (const float*)d_in[8];   // (R, 64)
    const float* b_nb = (const float*)d_in[9];
    const float* w_s1 = (const float*)d_in[10];
    const float* g_s1 = (const float*)d_in[11];
    const float* b_s1 = (const float*)d_in[12];
    const float* w_s2 = (const float*)d_in[13];
    const float* b_s2 = (const float*)d_in[14];

    // Workspace layout (~20.3 MB total)
    char* ws = (char*)d_ws;
    bf16_t* hA  = (bf16_t*)(ws);                            //  8 MB
    bf16_t* hB  = (bf16_t*)(ws + (size_t)8  * 1024 * 1024); //  8 MB
    float*  sq  = (float*) (ws + (size_t)16 * 1024 * 1024); // 256 KB
    int*    idx = (int*)   (ws + (size_t)16 * 1024 * 1024 + 512 * 1024); // 4 MB

    stem_kernel<<<(Bn * Nn) / 256, 256, 0, stream>>>(
        x, w_t1, g_t1, b_t1, w_t2, g_t2, b_t2, hA, sq);

    bf16_t* hin = hA;
    bf16_t* hout = hB;
    for (int r = 0; r < Rr; ++r) {
        knn_kernel<<<Bn * (Nn / 16), 128, 0, stream>>>(hin, sq, idx);
        edgeconv_kernel<<<Bn * 8, 256, 0, stream>>>(
            hin, idx, w_nb + (size_t)r * Hh * 2 * Hh,
            g_nb + (size_t)r * Hh, b_nb + (size_t)r * Hh, hout, sq);
        bf16_t* t = hin; hin = hout; hout = t;
    }

    head_kernel<<<(Bn * Nn) / 256, 256, 0, stream>>>(
        hin, w_s1, g_s1, b_s1, w_s2, b_s2, (float*)d_out);
}
